// DeepSeekMoELayer_11106785427616
// MI455X (gfx1250) — matrix operand order repaired and checked by reading, not compiled, not run on hardware
//
#include <hip/hip_runtime.h>
#include <hip/hip_bf16.h>

// ---------------- CDNA5 WMMA / TDM types ----------------
typedef __attribute__((ext_vector_type(16))) __bf16 v16bf;
typedef __attribute__((ext_vector_type(8)))  float  v8f;
typedef __attribute__((ext_vector_type(4)))  unsigned int u32x4;
typedef __attribute__((ext_vector_type(8)))  int    i32x8;
typedef __attribute__((ext_vector_type(4)))  int    i32x4;

// Problem constants (match reference setup_inputs)
#define T_TOK 2048
#define D_DIM 1024
#define F_DIM 512
#define E_EXP 64
#define K_TOP 8

// ---------------- WMMA fragment loaders (bf16 source data) ----------------
// A-matrix 16x32 bf16 (ISA 7.12.2): lane l holds row m = l&15;
// a[i] = k base+i (i<8), a[8+i] = k base+16+i, base = ((l>>4)<<3) + koff.
// Two contiguous 16B reads per fragment.
__device__ __forceinline__ v16bf load_a_bf16(const __bf16* Arow, int koff) {
    v16bf a;
    const int l = threadIdx.x & 31;
    const int base = koff + ((l >> 4) << 3);
#pragma unroll
    for (int i = 0; i < 8; ++i) {
        a[i]     = Arow[base + i];
        a[8 + i] = Arow[base + 16 + i];
    }
    return a;
}

// B-matrix 32x16 bf16 from N-major transposed weights Bt[n][k] (ldk = K):
// lane l holds col n; b[i] = Bt[n][kbase + i], b[8+i] = Bt[n][kbase + 16 + i], kbase = koff + ((l>>4)<<3).
// two 16B contiguous runs per lane, zero conversions.
__device__ __forceinline__ v16bf load_b_bf16t(const __bf16* Bt, int ldk, int n, int koff) {
    v16bf b;
    const int l = threadIdx.x & 31;
    const __bf16* r = Bt + (size_t)n * ldk + koff + ((l >> 4) << 3);
#pragma unroll
    for (int i = 0; i < 8; ++i) { b[i] = r[i]; b[8 + i] = r[16 + i]; }
    return b;
}

__device__ __forceinline__ v8f wmma_bf16(v16bf a, v16bf b, v8f c) {
    return __builtin_amdgcn_wmma_f32_16x16x32_bf16(
        false, a, false, b, (short)0, c, false, false);
}

__device__ __forceinline__ float silu(float x) { return x / (1.0f + __expf(-x)); }

// ---------------- TDM gather: 16 scattered bf16 rows -> LDS ----------------
// D# per CDNA5 ISA ch.8; gather_mode=1, 16-bit indices, data_size=2B.
__device__ __forceinline__ void tdm_gather16_bf16(unsigned lds_byte_off,
                                                  const __bf16* gbase,
                                                  unsigned row_elems,
                                                  unsigned row_stride_elems,
                                                  unsigned n_rows,
                                                  const int* toks /*16 ids*/) {
    const unsigned long long ga = (unsigned long long)(uintptr_t)gbase;
    u32x4 g0;
    g0[0] = 1u | (1u << 31);                         // count=1, gather_mode=1, 16-bit idx
    g0[1] = lds_byte_off;                            // LDS destination (bytes)
    g0[2] = (unsigned)(ga & 0xffffffffu);            // global_addr[31:0]
    g0[3] = (unsigned)((ga >> 32) & 0x1ffffffu) | (2u << 30);  // addr[56:32] | type=2
    i32x8 g1;
    g1[0] = (int)(1u << 16);                         // workgroup_mask=0, data_size=2B
    g1[1] = (int)((row_elems & 0xffffu) << 16);      // tensor_dim0[15:0]
    g1[2] = (int)(((row_elems >> 16) & 0xffffu) | ((T_TOK & 0xffffu) << 16));
    g1[3] = (int)((row_elems & 0xffffu) << 16);      // tile_dim0
    g1[4] = (int)(n_rows & 0xffffu);                 // tile_dim1 = #valid indices
    g1[5] = (int)row_stride_elems;                   // tensor_dim0_stride[31:0]
    g1[6] = 0;
    g1[7] = 0;
    i32x4 g2, g3;
#pragma unroll
    for (int i = 0; i < 4; ++i) {
        g2[i] = (toks[2 * i] & 0xffff) | ((toks[2 * i + 1] & 0xffff) << 16);
        g3[i] = (toks[8 + 2 * i] & 0xffff) | ((toks[8 + 2 * i + 1] & 0xffff) << 16);
    }
    i32x8 g4 = {0, 0, 0, 0, 0, 0, 0, 0};
    __builtin_amdgcn_tensor_load_to_lds(g0, g1, g2, g3, g4, 0);
}

// ---------------- preprocessing: fp32 -> bf16 ----------------
// Elementwise convert (x and activations), 4 elements/thread.
__global__ __launch_bounds__(256) void cvt_bf16_kernel(
    const float* __restrict__ src, __bf16* __restrict__ dst, int n4) {
    const int i = (blockIdx.x * 256 + threadIdx.x);
    if (i < n4) {
        const float4 f = *(const float4*)(src + 4 * (size_t)i);
        __bf16* d = dst + 4 * (size_t)i;
        d[0] = (__bf16)f.x; d[1] = (__bf16)f.y;
        d[2] = (__bf16)f.z; d[3] = (__bf16)f.w;
    }
}

// Transpose-convert W[K][N] fp32 -> Wt[N][K] bf16 via 32x32 LDS tile.
// blockIdx.z = batch (expert) index.
__global__ __launch_bounds__(256) void transpose_cvt_kernel(
    const float* __restrict__ W, __bf16* __restrict__ Wt, int K, int N) {
    __shared__ float tile[32][33];
    const size_t moff = (size_t)blockIdx.z * K * N;
    const int k0 = blockIdx.y << 5, n0 = blockIdx.x << 5;
    const int tx = threadIdx.x & 31, ty = threadIdx.x >> 5;  // 32 x 8
#pragma unroll
    for (int i = ty; i < 32; i += 8)
        tile[i][tx] = W[moff + (size_t)(k0 + i) * N + n0 + tx];
    __syncthreads();
#pragma unroll
    for (int i = ty; i < 32; i += 8)
        Wt[moff + (size_t)(n0 + i) * K + k0 + tx] = (__bf16)tile[tx][i];
}

// ---------------- workspace init ----------------
__global__ void zero_cnt_kernel(int* cnt) {
    if (threadIdx.x < E_EXP) cnt[threadIdx.x] = 0;
}

// ---------------- routing: top-8 + per-expert token lists ----------------
__global__ __launch_bounds__(64) void route_kernel(
    const float* __restrict__ x, const float* __restrict__ cent,
    const float* __restrict__ bias, int* cnt, int* tok_list, float* w_list) {
    __shared__ float xs[D_DIM];
    __shared__ float aff[E_EXP];
    const int t = blockIdx.x;
    const float* xr = x + (size_t)t * D_DIM;
    for (int i = threadIdx.x; i < D_DIM; i += 64) xs[i] = xr[i];
    __syncthreads();
    const int e = threadIdx.x;
    if (e < E_EXP) {
        const float* cr = cent + (size_t)e * D_DIM;
        float s = 0.f;
        for (int i = 0; i < D_DIM; ++i) s = fmaf(xs[i], cr[i], s);
        aff[e] = s;
    }
    __syncthreads();
    if (threadIdx.x == 0) {
        bool used[E_EXP];
        for (int i = 0; i < E_EXP; ++i) used[i] = false;
        float w[K_TOP]; int id[K_TOP];
        float wsum = 0.f;
        for (int k = 0; k < K_TOP; ++k) {
            float best = -1e30f; int bi = 0;
            for (int j = 0; j < E_EXP; ++j) {
                if (!used[j]) {
                    const float vv = aff[j] + bias[j];
                    if (vv > best) { best = vv; bi = j; }
                }
            }
            used[bi] = true;
            id[k] = bi;
            const float sw = 1.f / (1.f + __expf(-aff[bi]));
            w[k] = sw; wsum += sw;
        }
        const float inv = 1.f / (wsum + 1e-8f);
        for (int k = 0; k < K_TOP; ++k) {
            const int ex = id[k];
            const int p = atomicAdd(&cnt[ex], 1);
            tok_list[ex * T_TOK + p] = t;
            w_list[ex * T_TOK + p] = w[k] * inv;
        }
    }
}

// ---------------- wave-strip GEMM (shared expert): 16x64 per wave ----------------
// A: bf16 row-major [M][K]; Bt: bf16 N-major [N][K]; out fp32 or bf16.
template <bool SILU, bool OUT_BF16>
__global__ __launch_bounds__(32) void gemm_bf16_kernel(
    const __bf16* __restrict__ A, const __bf16* __restrict__ Bt,
    float* __restrict__ Cf, __bf16* __restrict__ Cb, int M, int N, int K) {
    const int strips_n = N >> 6;
    const int tm = blockIdx.x / strips_n;
    const int tn = blockIdx.x - tm * strips_n;
    const int l = threadIdx.x;
    const int m = (tm << 4) + (l & 15);
    const int nb = (tn << 6) + (l & 15);
    const __bf16* Arow = A + (size_t)m * K;
    v8f acc0 = {}, acc1 = {}, acc2 = {}, acc3 = {};
    for (int k = 0; k < K; k += 32) {
        if (k + 32 < K)
            __builtin_prefetch(Bt + (size_t)nb * K + k + 32, 0, 1);  // global_prefetch_b8
        const v16bf a = load_a_bf16(Arow, k);        // one A fragment feeds 4 WMMAs
        acc0 = wmma_bf16(a, load_b_bf16t(Bt, K, nb,      k), acc0);
        acc1 = wmma_bf16(a, load_b_bf16t(Bt, K, nb + 16, k), acc1);
        acc2 = wmma_bf16(a, load_b_bf16t(Bt, K, nb + 32, k), acc2);
        acc3 = wmma_bf16(a, load_b_bf16t(Bt, K, nb + 48, k), acc3);
    }
    const int rbase = (tm << 4) + ((l >> 4) << 3);
    v8f accs[4] = {acc0, acc1, acc2, acc3};
#pragma unroll
    for (int j = 0; j < 4; ++j) {
#pragma unroll
        for (int v = 0; v < 8; ++v) {
            float y = accs[j][v];
            if (SILU) y = silu(y);
            if (OUT_BF16)
                Cb[(size_t)(rbase + v) * N + nb + 16 * j] = (__bf16)y;
            else
                Cf[(size_t)(rbase + v) * N + nb + 16 * j] = y;
        }
    }
}

// ---------------- fused routed expert ----------------
// Per block: (expert e, 16-token tile). One TDM gather stages the full bf16
// token rows (16 x 1024 = 32 KB) in LDS; pass1 h = silu(Xg@Wup) -> bf16 LDS;
// pass2 y = h@Wdown with weighted atomic scatter into out.
__global__ __launch_bounds__(256) void moe_expert_kernel(
    const __bf16* __restrict__ xb, const __bf16* __restrict__ Wut,
    const __bf16* __restrict__ Wdt, const int* __restrict__ cnt,
    const int* __restrict__ tok_list, const float* __restrict__ w_list,
    float* __restrict__ out) {
    const int tiles_per_e = T_TOK >> 4;             // 128
    const int e    = blockIdx.x / tiles_per_e;
    const int tile = blockIdx.x - e * tiles_per_e;
    const int ntok = cnt[e];
    if ((tile << 4) >= ntok) return;

    __shared__ __bf16 xgb[16 * D_DIM];              // 32 KB gathered-X (TDM dest)
    __shared__ __bf16 hsb[16 * F_DIM];              // 16 KB bf16 activations
    __shared__ int   toks[16];
    __shared__ float wts[16];

    if (threadIdx.x < 16) {
        const int p = (tile << 4) + threadIdx.x;
        const bool valid = p < ntok;
        toks[threadIdx.x] = valid ? tok_list[e * T_TOK + p] : tok_list[e * T_TOK];
        wts[threadIdx.x]  = valid ? w_list[e * T_TOK + p] : 0.f;
    }
    __syncthreads();
    if (threadIdx.x < 32) {                         // wave 0 drives the TDM gather
        tdm_gather16_bf16((unsigned)(uintptr_t)(&xgb[0]), xb,
                          D_DIM, D_DIM, 16, toks);
        __builtin_amdgcn_s_wait_tensorcnt(0);
    }
    __syncthreads();                                // gathered rows visible to all

    const int wave = threadIdx.x >> 5;
    const int l = threadIdx.x & 31;
    const int mloc = l & 15;

    // ---- Pass 1: h[16 x F] = silu(Xg @ Wup[e]) ----
    const __bf16* Wut_e = Wut + (size_t)e * F_DIM * D_DIM;  // [F][D] n-major
    {
        const int n0 = (wave << 6) + (l & 15);      // 8 waves x 64-wide strips = F
        const __bf16* Arow = &xgb[mloc * D_DIM];
        v8f acc0 = {}, acc1 = {}, acc2 = {}, acc3 = {};
        for (int k = 0; k < D_DIM; k += 32) {
            if (k + 32 < D_DIM)
                __builtin_prefetch(Wut_e + (size_t)n0 * D_DIM + k + 32, 0, 1);
            const v16bf a = load_a_bf16(Arow, k);
            acc0 = wmma_bf16(a, load_b_bf16t(Wut_e, D_DIM, n0,      k), acc0);
            acc1 = wmma_bf16(a, load_b_bf16t(Wut_e, D_DIM, n0 + 16, k), acc1);
            acc2 = wmma_bf16(a, load_b_bf16t(Wut_e, D_DIM, n0 + 32, k), acc2);
            acc3 = wmma_bf16(a, load_b_bf16t(Wut_e, D_DIM, n0 + 48, k), acc3);
        }
        v8f accs[4] = {acc0, acc1, acc2, acc3};
#pragma unroll
        for (int j = 0; j < 4; ++j) {
#pragma unroll
            for (int v = 0; v < 8; ++v) {
                const int row = v + ((l >> 4) << 3);
                hsb[row * F_DIM + n0 + 16 * j] = (__bf16)silu(accs[j][v]);
            }
        }
    }
    __syncthreads();

    // ---- Pass 2: y[16 x D] = h @ Wdown[e]; weighted atomic combine ----
    const __bf16* Wdt_e = Wdt + (size_t)e * D_DIM * F_DIM;  // [D][F] n-major
    const __bf16* Hrow = &hsb[mloc * F_DIM];
#pragma unroll
    for (int g = 0; g < 2; ++g) {                   // 2 x (8 waves x 64-wide) = D
        const int n0 = ((wave + 8 * g) << 6) + (l & 15);
        v8f acc0 = {}, acc1 = {}, acc2 = {}, acc3 = {};
        for (int k = 0; k < F_DIM; k += 32) {
            if (k + 32 < F_DIM)
                __builtin_prefetch(Wdt_e + (size_t)n0 * F_DIM + k + 32, 0, 1);
            const v16bf a = load_a_bf16(Hrow, k);   // contiguous bf16 LDS reads
            acc0 = wmma_bf16(a, load_b_bf16t(Wdt_e, F_DIM, n0,      k), acc0);
            acc1 = wmma_bf16(a, load_b_bf16t(Wdt_e, F_DIM, n0 + 16, k), acc1);
            acc2 = wmma_bf16(a, load_b_bf16t(Wdt_e, F_DIM, n0 + 32, k), acc2);
            acc3 = wmma_bf16(a, load_b_bf16t(Wdt_e, F_DIM, n0 + 48, k), acc3);
        }
        v8f accs[4] = {acc0, acc1, acc2, acc3};
#pragma unroll
        for (int j = 0; j < 4; ++j) {
#pragma unroll
            for (int v = 0; v < 8; ++v) {
                const int row = v + ((l >> 4) << 3);
                const float wgt = wts[row];
                if (wgt != 0.f)
                    atomicAdd(&out[(size_t)toks[row] * D_DIM + n0 + 16 * j],
                              wgt * accs[j][v]);
            }
        }
    }
}

// ---------------- host launcher ----------------
extern "C" void kernel_launch(void* const* d_in, const int* in_sizes, int n_in,
                              void* d_out, int out_size, void* d_ws, size_t ws_size,
                              hipStream_t stream) {
    (void)in_sizes; (void)n_in; (void)out_size; (void)ws_size;
    const float* x         = (const float*)d_in[0];  // [2,1024,1024]
    const float* centroids = (const float*)d_in[1];  // [64,1024]
    const float* biases    = (const float*)d_in[2];  // [64]
    const float* Ws_up     = (const float*)d_in[3];  // [1024,512]
    const float* Ws_down   = (const float*)d_in[4];  // [512,1024]
    const float* W_up      = (const float*)d_in[5];  // [64,1024,512]
    const float* W_down    = (const float*)d_in[6];  // [64,512,1024]
    float* out = (float*)d_out;                      // [2,1024,1024]

    // workspace layout (bytes)
    char* ws = (char*)d_ws;
    size_t off = 0;
    int*    cnt      = (int*)(ws + off);  off += 1024;
    int*    tok_list = (int*)(ws + off);  off += (size_t)E_EXP * T_TOK * 4;
    float*  w_list   = (float*)(ws + off); off += (size_t)E_EXP * T_TOK * 4;
    __bf16* xbb      = (__bf16*)(ws + off); off += (size_t)T_TOK * D_DIM * 2;
    __bf16* tmpb     = (__bf16*)(ws + off); off += (size_t)T_TOK * F_DIM * 2;
    __bf16* Wsu_t    = (__bf16*)(ws + off); off += (size_t)F_DIM * D_DIM * 2;
    __bf16* Wsd_t    = (__bf16*)(ws + off); off += (size_t)D_DIM * F_DIM * 2;
    __bf16* Wut      = (__bf16*)(ws + off); off += (size_t)E_EXP * F_DIM * D_DIM * 2;
    __bf16* Wdt      = (__bf16*)(ws + off); off += (size_t)E_EXP * D_DIM * F_DIM * 2;

    // 1) routing (builds per-expert token lists)
    zero_cnt_kernel<<<1, 64, 0, stream>>>(cnt);
    route_kernel<<<T_TOK, 64, 0, stream>>>(x, centroids, biases, cnt, tok_list, w_list);

    // 2) one-time bf16 preprocessing: x elementwise; weights transpose to N-major
    {
        const int n4 = (T_TOK * D_DIM) / 4;
        cvt_bf16_kernel<<<(n4 + 255) / 256, 256, 0, stream>>>(x, xbb, n4);
        dim3 g_su(F_DIM / 32, D_DIM / 32, 1);
        transpose_cvt_kernel<<<g_su, 256, 0, stream>>>(Ws_up, Wsu_t, D_DIM, F_DIM);
        dim3 g_sd(D_DIM / 32, F_DIM / 32, 1);
        transpose_cvt_kernel<<<g_sd, 256, 0, stream>>>(Ws_down, Wsd_t, F_DIM, D_DIM);
        dim3 g_up(F_DIM / 32, D_DIM / 32, E_EXP);
        transpose_cvt_kernel<<<g_up, 256, 0, stream>>>(W_up, Wut, D_DIM, F_DIM);
        dim3 g_dn(D_DIM / 32, F_DIM / 32, E_EXP);
        transpose_cvt_kernel<<<g_dn, 256, 0, stream>>>(W_down, Wdt, F_DIM, D_DIM);
    }

    // 3) shared expert: tmpb = silu(xb @ Wsu_t) [bf16]; out = tmpb @ Wsd_t [fp32]
    {
        const int blocks_up = (T_TOK / 16) * (F_DIM / 64);    // 1024
        gemm_bf16_kernel<true, true><<<blocks_up, 32, 0, stream>>>(
            xbb, Wsu_t, nullptr, tmpb, T_TOK, F_DIM, D_DIM);
        const int blocks_dn = (T_TOK / 16) * (D_DIM / 64);    // 2048
        gemm_bf16_kernel<false, false><<<blocks_dn, 32, 0, stream>>>(
            tmpb, Wsd_t, out, nullptr, T_TOK, D_DIM, F_DIM);
    }

    // 4) routed experts: TDM-gathered fused up->silu->down, atomic combine
    {
        const int blocks = E_EXP * (T_TOK / 16);              // 8192
        moe_expert_kernel<<<blocks, 256, 0, stream>>>(xbb, Wut, Wdt, cnt,
                                                      tok_list, w_list, out);
    }
}
